// RelationNetwork_60722247631026
// MI455X (gfx1250) — compile-verified, matrix operand order checked by reading, not run on hardware
//
#include <hip/hip_runtime.h>
#include <hip/hip_bf16.h>

typedef __attribute__((ext_vector_type(16))) __bf16 v16bf;
typedef __attribute__((ext_vector_type(8)))  __bf16 v8bf;
typedef __attribute__((ext_vector_type(8)))  float  v8f;

#define NPTS        512
#define FDIM        64
#define RDIM        128
#define NB          4
#define PAIRS       261632          // 512*511
#define TILES_PB    16352           // PAIRS/16
#define TOT_TILES   65408           // NB*TILES_PB
#define TILES_PER_WAVE 16
#define WAVES_PER_BLOCK 8
#define MAIN_BLOCKS (TOT_TILES / (TILES_PER_WAVE * WAVES_PER_BLOCK))   // 511

// ---------------------------------------------------------------------------
// Kernel 1: A'[b,n,k] = feat[b,n,:] @ W1[0:64, k] + b1[k]
//           C [b,n,k] = feat[b,n,:] @ W1[64:128, k]
// One thread per (b,n,k): 2048*128 = 262144 threads; trivial FLOPs.
// ---------------------------------------------------------------------------
__global__ __launch_bounds__(256)
void relnet_precompute(const float* __restrict__ feat,
                       const float* __restrict__ W1,
                       const float* __restrict__ b1,
                       float* __restrict__ Ap,
                       float* __restrict__ Cp) {
    int tid = blockIdx.x * 256 + threadIdx.x;        // 0 .. 262143
    int bn  = tid >> 7;                              // row in [0, B*N)
    int k   = tid & 127;
    const float* __restrict__ f = feat + bn * FDIM;
    float s1 = b1[k];
    float s2 = 0.0f;
#pragma unroll 8
    for (int q = 0; q < FDIM; ++q) {
        float v = f[q];
        s1 = fmaf(v, W1[q * RDIM + k],          s1);
        s2 = fmaf(v, W1[(q + FDIM) * RDIM + k], s2);
    }
    Ap[tid] = s1;     // tid == bn*128 + k
    Cp[tid] = s2;
}

// ---------------------------------------------------------------------------
// Kernel 2: for each 16-pair tile:
//   h1 = relu(A'[i] + C[j])            (fp32, cvt -> bf16 A-operand)
//   h2 = relu(h1 @ W2 + b2)            (v_wmma_f32_16x16x32_bf16, f32 accum)
//   out = h2 @ W3 + b3                 (per-lane FMA + shfl_xor butterfly)
// W2 staged once per block into LDS in exact B-operand lane order
// (coalesced global read, scattered LDS write); the compiler then hoists the
// B operands into high VGPRs (1024-VGPR file) and keeps them for all tiles.
// ---------------------------------------------------------------------------
__global__ __launch_bounds__(256)
void relnet_main(const float* __restrict__ Ap,
                 const float* __restrict__ Cp,
                 const float* __restrict__ W2,
                 const float* __restrict__ b2,
                 const float* __restrict__ W3,
                 const float* __restrict__ b3,
                 float* __restrict__ out) {
    __shared__ __attribute__((aligned(64))) __bf16 sBpack[32 * 32 * 16]; // 32KB
    __shared__ float sB2[RDIM];
    __shared__ float sW3[RDIM];
    __shared__ float sOut[WAVES_PER_BLOCK * 16];

    const int tid = threadIdx.x;

    // ---- stage W2 -> LDS in B-operand order.
    // Iterate W2 linearly (coalesced global loads); scatter into
    // sBpack[((kk*8+nt)*32 + lane)*16 + e] where for element (k, n):
    //   kk = k>>5, krem = k&31, hi = (krem>>3)&1, e = (krem&7) + ((krem>>4)<<3)
    //   nt = n>>4, lane = (n&15) + 16*hi
    for (int s = tid; s < RDIM * RDIM; s += 256) {
        int k = s >> 7;                 // W2 row (K index)
        int n = s & 127;                // W2 col (N index)
        int kk   = k >> 5;
        int krem = k & 31;
        int hi2  = (krem >> 3) & 1;
        int e    = (krem & 7) + ((krem >> 4) << 3);
        int nt   = n >> 4;
        int ln   = (n & 15) + (hi2 << 4);
        sBpack[((kk * 8 + nt) * 32 + ln) * 16 + e] = (__bf16)W2[s];
    }
    if (tid < RDIM) {
        sB2[tid] = b2[tid];
        sW3[tid] = W3[tid];
    }
    __syncthreads();

    const int wv      = tid >> 5;
    const int lane    = tid & 31;
    const int lanelow = lane & 15;
    const int hi      = lane >> 4;
    const float b3v   = b3[0];

    for (int it = 0; it < TILES_PER_WAVE; ++it) {
        int t  = blockIdx.x * (TILES_PER_WAVE * WAVES_PER_BLOCK) + it * WAVES_PER_BLOCK + wv;
        int b  = t / TILES_PB;
        int tp = t - b * TILES_PB;
        int p0 = tp * 16;

        // pair (i, j) for this lane's row (row m = lanelow)
        int p = p0 + lanelow;
        int i = p / (NPTS - 1);
        int r = p - i * (NPTS - 1);
        int j = r + ((r >= i) ? 1 : 0);

        const float* __restrict__ rowA = Ap + ((b * NPTS + i) << 7);
        const float* __restrict__ rowC = Cp + ((b * NPTS + j) << 7);

        v8f acc[8];
#pragma unroll
        for (int nt = 0; nt < 8; ++nt) acc[nt] = (v8f)0.0f;

#pragma unroll
        for (int kk = 0; kk < 4; ++kk) {
            // A-operand: lane holds row lanelow, K = {kk*32+8*hi+0..7, kk*32+16+8*hi+0..7}
            const float* pa = rowA + kk * 32 + 8 * hi;
            const float* pc = rowC + kk * 32 + 8 * hi;
            v8f aLo = *(const v8f*)(pa);
            v8f aHi = *(const v8f*)(pa + 16);
            v8f cLo = *(const v8f*)(pc);
            v8f cHi = *(const v8f*)(pc + 16);
            v8f hLo = __builtin_elementwise_max(aLo + cLo, (v8f)0.0f);  // relu(h1)
            v8f hHi = __builtin_elementwise_max(aHi + cHi, (v8f)0.0f);
            v8bf bLo = __builtin_convertvector(hLo, v8bf);
            v8bf bHi = __builtin_convertvector(hHi, v8bf);
            v16bf av = __builtin_shufflevector(bLo, bHi,
                0, 1, 2, 3, 4, 5, 6, 7, 8, 9, 10, 11, 12, 13, 14, 15);

#pragma unroll
            for (int nt = 0; nt < 8; ++nt) {
                v16bf bv = *(const v16bf*)(&sBpack[((kk * 8 + nt) * 32 + lane) * 16]);
                acc[nt] = __builtin_amdgcn_wmma_f32_16x16x32_bf16(
                    false, av, false, bv, (short)0, acc[nt], false, false);
            }
        }

        // ---- epilogue: relu(acc + b2[n]) dotted with W3[n], reduce over n
        float part[8];
#pragma unroll
        for (int v = 0; v < 8; ++v) part[v] = 0.0f;
#pragma unroll
        for (int nt = 0; nt < 8; ++nt) {
            float b2v = sB2[nt * 16 + lanelow];
            float w3v = sW3[nt * 16 + lanelow];
#pragma unroll
            for (int v = 0; v < 8; ++v) {
                float h = fmaxf(acc[nt][v] + b2v, 0.0f);   // relu(h2)
                part[v] = fmaf(h, w3v, part[v]);
            }
        }
        // butterfly over the 16 lanes sharing a D-matrix row group
#pragma unroll
        for (int v = 0; v < 8; ++v) {
#pragma unroll
            for (int s = 1; s < 16; s <<= 1)
                part[v] += __shfl_xor(part[v], s, 32);
        }
        if (lanelow == 0) {
#pragma unroll
            for (int v = 0; v < 8; ++v)
                sOut[wv * 16 + 8 * hi + v] = part[v];
        }
        if (lane < 16)
            out[b * PAIRS + p0 + lane] = sOut[wv * 16 + lane] + b3v;
    }
}

extern "C" void kernel_launch(void* const* d_in, const int* in_sizes, int n_in,
                              void* d_out, int out_size, void* d_ws, size_t ws_size,
                              hipStream_t stream) {
    (void)in_sizes; (void)n_in; (void)out_size; (void)ws_size;
    const float* feat = (const float*)d_in[0];
    const float* W1   = (const float*)d_in[1];
    const float* b1   = (const float*)d_in[2];
    const float* W2   = (const float*)d_in[3];
    const float* b2   = (const float*)d_in[4];
    const float* W3   = (const float*)d_in[5];
    const float* b3   = (const float*)d_in[6];
    float* out = (float*)d_out;

    float* Ap = (float*)d_ws;                 // [B*N, 128] fp32
    float* Cp = Ap + NB * NPTS * RDIM;        // [B*N, 128] fp32  (2 MB total)

    relnet_precompute<<<(NB * NPTS * RDIM) / 256, 256, 0, stream>>>(feat, W1, b1, Ap, Cp);
    relnet_main<<<MAIN_BLOCKS, 256, 0, stream>>>(Ap, Cp, W2, b2, W3, b3, out);
}
